// Attention_27101243637977
// MI455X (gfx1250) — matrix operand order repaired and checked by reading, not compiled, not run on hardware
//
#include <hip/hip_runtime.h>
#include <hip/hip_bf16.h>

typedef __attribute__((ext_vector_type(16))) _Float16 v16h;
typedef __attribute__((ext_vector_type(8)))  _Float16 v8h;
typedef __attribute__((ext_vector_type(8)))  float    v8f;

#define BATCH  2
#define CHN    64
#define DDIM   32
#define HDIM   64
#define WDIM   64
#define HEADS  8
#define NTOK   2048
#define FDIM   512
#define KEYBLK 128
#define TOT    (1u << 24)   /* B*C*D*H*W = 2*64*32*64*64 = 2^24 */
#define SPAT   (1 << 17)    /* D*H*W = 131072 */

__device__ __forceinline__ size_t idx5(int b, int c, int d, int h, int w) {
    return ((((size_t)b * CHN + c) * DDIM + d) * HDIM + h) * WDIM + w;
}

// ---------------------------------------------------------------- depthwise 3x3x3
__global__ __launch_bounds__(256) void k_dwconv333(
    const float* __restrict__ in, const float* __restrict__ w,
    const float* __restrict__ bias, float* __restrict__ out) {
    unsigned i = blockIdx.x * 256u + threadIdx.x;
    if (i >= TOT) return;
    int ww = i & 63, hh = (i >> 6) & 63, dd = (i >> 12) & 31;
    int ch = (i >> 17) & 63, b = i >> 23;
    float acc = bias[ch];
    const float* wp = w + ch * 27;
    for (int kd = -1; kd <= 1; ++kd) {
        int d2 = dd + kd; if (d2 < 0 || d2 >= DDIM) continue;
        for (int kh = -1; kh <= 1; ++kh) {
            int h2 = hh + kh; if (h2 < 0 || h2 >= HDIM) continue;
            for (int kw = -1; kw <= 1; ++kw) {
                int w2 = ww + kw; if (w2 < 0 || w2 >= WDIM) continue;
                acc += wp[(kd + 1) * 9 + (kh + 1) * 3 + (kw + 1)] * in[idx5(b, ch, d2, h2, w2)];
            }
        }
    }
    out[i] = acc;
}

// ---------------------------------------------------------------- depthwise 1-D (3 taps), axis 0=D 1=H 2=W
__global__ __launch_bounds__(256) void k_dwconv1d(
    const float* __restrict__ in, const float* __restrict__ w,
    const float* __restrict__ bias, float* __restrict__ out,
    int axis, int accumulate) {
    unsigned i = blockIdx.x * 256u + threadIdx.x;
    if (i >= TOT) return;
    int ww = i & 63, hh = (i >> 6) & 63, dd = (i >> 12) & 31;
    int ch = (i >> 17) & 63, b = i >> 23;
    float acc = bias[ch];
    const float* wp = w + ch * 3;
    for (int t = -1; t <= 1; ++t) {
        int d2 = dd, h2 = hh, w2 = ww;
        if (axis == 0) d2 += t; else if (axis == 1) h2 += t; else w2 += t;
        if (d2 >= 0 && d2 < DDIM && h2 >= 0 && h2 < HDIM && w2 >= 0 && w2 < WDIM)
            acc += wp[t + 1] * in[idx5(b, ch, d2, h2, w2)];
    }
    if (accumulate) out[i] += acc; else out[i] = acc;
}

// ---------------------------------------------------------------- InstanceNorm: stats then apply (in place)
__global__ __launch_bounds__(256) void k_inorm_stats(const float* __restrict__ x,
                                                     float* __restrict__ stats) {
    int bc = blockIdx.x;                       // 0..127
    const float* p = x + (size_t)bc * SPAT;
    float s = 0.f, ss = 0.f;
    for (int i = threadIdx.x; i < SPAT; i += 256) { float v = p[i]; s += v; ss += v * v; }
    __shared__ float rs_[256], rss_[256];
    int tid = threadIdx.x;
    rs_[tid] = s; rss_[tid] = ss; __syncthreads();
    for (int o = 128; o > 0; o >>= 1) {
        if (tid < o) { rs_[tid] += rs_[tid + o]; rss_[tid] += rss_[tid + o]; }
        __syncthreads();
    }
    if (tid == 0) {
        float m = rs_[0] / (float)SPAT;
        float var = rss_[0] / (float)SPAT - m * m;
        stats[2 * bc] = m;
        stats[2 * bc + 1] = rsqrtf(var + 1e-5f);
    }
}

__global__ __launch_bounds__(256) void k_inorm_apply(float* __restrict__ x,
                                                     const float* __restrict__ stats) {
    unsigned i = blockIdx.x * 256u + threadIdx.x;
    if (i >= TOT) return;
    int bc = i >> 17;
    x[i] = (x[i] - stats[2 * bc]) * stats[2 * bc + 1];
}

// ---------------------------------------------------------------- QKV pack: 5D f32 -> f16 [bh][n][f] (or transposed [bh][f][n])
// arr: 0 = dh (n = d*H+h, f = w*8+c) ; 1 = dw (n = d*W+w, f = h*8+c)
__global__ __launch_bounds__(128) void k_qkv_pack(
    const float* __restrict__ src, _Float16* __restrict__ dst,
    int arr, int doNorm, int transOut) {
    int blk = blockIdx.x;
    int n = blk & (NTOK - 1);
    int bh = blk >> 11;
    int b = bh >> 3, hd = bh & 7;
    int tid = threadIdx.x;
    float vals[4]; int fs[4];
    float ssq = 0.f;
#pragma unroll
    for (int j = 0; j < 4; ++j) {
        int f = tid + j * 128;
        int dd, hh, ww, cc;
        if (arr == 0) { dd = n >> 6; hh = n & 63; ww = f >> 3; cc = f & 7; }
        else          { dd = n >> 6; ww = n & 63; hh = f >> 3; cc = f & 7; }
        float v = src[idx5(b, hd * 8 + cc, dd, hh, ww)];
        vals[j] = v; fs[j] = f; ssq += v * v;
    }
    float scale = 1.f;
    if (doNorm) {
        __shared__ float red[128];
        red[tid] = ssq; __syncthreads();
        for (int o = 64; o > 0; o >>= 1) {
            if (tid < o) red[tid] += red[tid + o];
            __syncthreads();
        }
        scale = 1.f / fmaxf(sqrtf(red[0]), 1e-12f);
    }
    _Float16* dp = dst + (size_t)bh * NTOK * FDIM;
#pragma unroll
    for (int j = 0; j < 4; ++j) {
        size_t o = transOut ? ((size_t)fs[j] * NTOK + n) : ((size_t)n * FDIM + fs[j]);
        dp[o] = (_Float16)(vals[j] * scale);
    }
}

// ---------------------------------------------------------------- flash attention (no-max softmax: logits in [-1,1])
// Block = 8 waves, one 16-row Q tile. Wave w: S for keys [kb+16w,16), PV for cols [64w,64).
__global__ __launch_bounds__(256) void k_attn(
    const _Float16* __restrict__ Q,   // [bh][n][f]
    const _Float16* __restrict__ K,   // [bh][n][f]
    const _Float16* __restrict__ VT,  // [bh][f][n]
    const float* __restrict__ resid,  // out_k 5D
    float* __restrict__ outbuf,       // 5D, fully overwritten
    int arr) {
    __shared__ __align__(16) _Float16 sQ[16 * FDIM];    // 16 KB
    __shared__ __align__(16) _Float16 sP[16 * KEYBLK];  // 4 KB

    int blk = blockIdx.x;
    int rowTile = blk & (NTOK / 16 - 1);
    int bh = blk >> 7;                       // 0..15
    int b = bh >> 3, hd = bh & 7;
    int n0 = rowTile * 16;

    const _Float16* Qb  = Q  + (size_t)bh * NTOK * FDIM;
    const _Float16* Kb  = K  + (size_t)bh * NTOK * FDIM;
    const _Float16* VTb = VT + (size_t)bh * NTOK * FDIM;

    int tid = threadIdx.x;
    int wave = tid >> 5, lane = tid & 31;
    int mrow = lane & 15, hi = lane >> 4;

    // cooperative load of Q tile (coalesced b128)
    {
        const v8h* src = (const v8h*)(Qb + (size_t)n0 * FDIM);
        v8h* dst = (v8h*)sQ;
        for (int i = tid; i < 16 * FDIM / 8; i += 256) dst[i] = src[i];
    }
    __syncthreads();

    v8f zero = {0.f, 0.f, 0.f, 0.f, 0.f, 0.f, 0.f, 0.f};
    v8f acc[4] = {zero, zero, zero, zero};
    float rsum[8] = {0.f, 0.f, 0.f, 0.f, 0.f, 0.f, 0.f, 0.f};
    int colBase = wave * 64;

    for (int kb = 0; kb < NTOK; kb += KEYBLK) {
        // ---- phase A: S[16x16] for this wave's 16 keys, over F=512
        int key0 = kb + wave * 16;
        v8f s = zero;
        const _Float16* Kp = Kb + (size_t)(key0 + mrow) * FDIM + hi * 8;
        const _Float16* Qp = sQ + mrow * FDIM + hi * 8;
#pragma unroll 4
        for (int ks = 0; ks < FDIM; ks += 32) {
            v16h a, bf;
            *(v8h*)&a       = *(const v8h*)(Qp + ks);
            *((v8h*)&a + 1) = *(const v8h*)(Qp + ks + 16);
            *(v8h*)&bf       = *(const v8h*)(Kp + ks);
            *((v8h*)&bf + 1) = *(const v8h*)(Kp + ks + 16);
            s = __builtin_amdgcn_wmma_f32_16x16x32_f16(false, a, false, bf,
                                                       (short)0, s, false, false);
        }
        __syncthreads();   // previous-iteration P readers done
#pragma unroll
        for (int r = 0; r < 8; ++r) {
            float e = __expf(s[r]);
            rsum[r] += e;
            sP[(r + 8 * hi) * KEYBLK + wave * 16 + mrow] = (_Float16)e;
        }
        __syncthreads();

        // ---- phase B: O[16 x 64] += P[16 x 128] * V[128 x 64]
#pragma unroll
        for (int kc = 0; kc < 4; ++kc) {
            v16h pa;
            const _Float16* Pp = sP + mrow * KEYBLK + kc * 32 + hi * 8;
            *(v8h*)&pa       = *(const v8h*)(Pp);
            *((v8h*)&pa + 1) = *(const v8h*)(Pp + 16);
#pragma unroll
            for (int ct = 0; ct < 4; ++ct) {
                const _Float16* Vp = VTb + (size_t)(colBase + ct * 16 + mrow) * NTOK
                                         + kb + kc * 32 + hi * 8;
                v16h vb;
                *(v8h*)&vb       = *(const v8h*)(Vp);
                *((v8h*)&vb + 1) = *(const v8h*)(Vp + 16);
                acc[ct] = __builtin_amdgcn_wmma_f32_16x16x32_f16(false, pa, false, vb,
                                                                 (short)0, acc[ct], false, false);
            }
        }
    }

    // reduce row sums across the 16 lanes sharing each row half
#pragma unroll
    for (int r = 0; r < 8; ++r) {
        float v = rsum[r];
        for (int off = 1; off < 16; off <<= 1) v += __shfl_xor(v, off, 32);
        rsum[r] = v;
    }

    // epilogue: softmax divide + scatter into 5D layout with residual (fused from_dh/from_dw)
#pragma unroll
    for (int ct = 0; ct < 4; ++ct) {
        int f = colBase + ct * 16 + mrow;
#pragma unroll
        for (int r = 0; r < 8; ++r) {
            int n = n0 + r + 8 * hi;
            float val = acc[ct][r] / rsum[r];
            int dd, hh, ww, cc;
            if (arr == 0) { dd = n >> 6; hh = n & 63; ww = f >> 3; cc = f & 7; }
            else          { dd = n >> 6; ww = n & 63; hh = f >> 3; cc = f & 7; }
            size_t o = idx5(b, hd * 8 + cc, dd, hh, ww);
            outbuf[o] = val + resid[o];
        }
    }
}

// ---------------------------------------------------------------- final grouped conv: 128->64 ch, 2 in-ch/group, 3x3x3
__global__ __launch_bounds__(256) void k_finalconv(
    const float* __restrict__ o1, const float* __restrict__ o2,
    const float* __restrict__ w, const float* __restrict__ bias,
    float* __restrict__ out) {
    unsigned i = blockIdx.x * 256u + threadIdx.x;
    if (i >= TOT) return;
    int ww = i & 63, hh = (i >> 6) & 63, dd = (i >> 12) & 31;
    int ch = (i >> 17) & 63, b = i >> 23;
    float acc = bias[ch];
    const float* wp = w + ch * 54;
    for (int ic = 0; ic < 2; ++ic) {
        int cc = ch * 2 + ic;                      // concat channel
        const float* src = (cc < 64) ? o1 : o2;
        int sc = cc & 63;
        for (int kd = -1; kd <= 1; ++kd) {
            int d2 = dd + kd; if (d2 < 0 || d2 >= DDIM) continue;
            for (int kh = -1; kh <= 1; ++kh) {
                int h2 = hh + kh; if (h2 < 0 || h2 >= HDIM) continue;
                for (int kw = -1; kw <= 1; ++kw) {
                    int w2 = ww + kw; if (w2 < 0 || w2 >= WDIM) continue;
                    acc += wp[ic * 27 + (kd + 1) * 9 + (kh + 1) * 3 + (kw + 1)]
                         * src[idx5(b, sc, d2, h2, w2)];
                }
            }
        }
    }
    out[i] = acc;
}

// ================================================================ host side
extern "C" void kernel_launch(void* const* d_in, const int* in_sizes, int n_in,
                              void* d_out, int out_size, void* d_ws, size_t ws_size,
                              hipStream_t stream) {
    (void)in_sizes; (void)n_in; (void)out_size; (void)ws_size;

    const float* x0 = (const float*)d_in[0];
    const float* x1 = (const float*)d_in[1];
    const float* x2 = (const float*)d_in[2];
    const float* wd  = (const float*)d_in[3],  *bd  = (const float*)d_in[4];
    const float* w01 = (const float*)d_in[5],  *b01 = (const float*)d_in[6];
    const float* w02 = (const float*)d_in[7],  *b02 = (const float*)d_in[8];
    const float* w03 = (const float*)d_in[9],  *b03 = (const float*)d_in[10];
    const float* w04 = (const float*)d_in[11], *b04 = (const float*)d_in[12];
    const float* w11 = (const float*)d_in[13], *b11 = (const float*)d_in[14];
    const float* w12 = (const float*)d_in[15], *b12 = (const float*)d_in[16];
    const float* w13 = (const float*)d_in[17], *b13 = (const float*)d_in[18];
    const float* w14 = (const float*)d_in[19], *b14 = (const float*)d_in[20];
    const float* w21 = (const float*)d_in[21], *b21 = (const float*)d_in[22];
    const float* w22 = (const float*)d_in[23], *b22 = (const float*)d_in[24];
    const float* w23 = (const float*)d_in[25], *b23 = (const float*)d_in[26];
    const float* w24 = (const float*)d_in[27], *b24 = (const float*)d_in[28];
    const float* wc  = (const float*)d_in[29], *bc  = (const float*)d_in[30];
    float* out = (float*)d_out;

    const size_t S = (size_t)TOT;   // 2^24 elements
    float* buf0 = (float*)d_ws;     // conv temp, later q1/k1 (f16)
    float* buf1 = buf0 + S;         // out1 -> out_k (residual, kept live)
    float* buf2 = buf1 + S;         // out2 -> out_q2, later k2/v2t (f16)
    float* buf3 = buf2 + S;         // out3 -> out_q1, later v1t/q2 (f16)
    float* buf4 = buf3 + S;         // conv temp, then attn1 output
    float* buf5 = buf4 + S;         // attn2 output
    float* stats = buf5 + S;        // 3 * 256 floats

    _Float16* q1  = (_Float16*)buf0;
    _Float16* k1  = (_Float16*)(buf0 + S / 2);
    _Float16* v1t = (_Float16*)buf3;
    _Float16* q2  = (_Float16*)(buf3 + S / 2);
    _Float16* k2  = (_Float16*)buf2;
    _Float16* v2t = (_Float16*)(buf2 + S / 2);

    dim3 eb(256), eg((TOT + 255) / 256);

    // ---- branch 0: out1 = W(H(t0)) + H(W(t0)), t0 = dw333(x0)
    k_dwconv333<<<eg, eb, 0, stream>>>(x0, wd, bd, buf0);
    k_dwconv1d<<<eg, eb, 0, stream>>>(buf0, w01, b01, buf4, 1, 0);
    k_dwconv1d<<<eg, eb, 0, stream>>>(buf4, w03, b03, buf1, 2, 0);
    k_dwconv1d<<<eg, eb, 0, stream>>>(buf0, w04, b04, buf4, 2, 0);
    k_dwconv1d<<<eg, eb, 0, stream>>>(buf4, w02, b02, buf1, 1, 1);
    // ---- branch 1: out2 (D/W chains)
    k_dwconv333<<<eg, eb, 0, stream>>>(x1, wd, bd, buf0);
    k_dwconv1d<<<eg, eb, 0, stream>>>(buf0, w11, b11, buf4, 0, 0);
    k_dwconv1d<<<eg, eb, 0, stream>>>(buf4, w13, b13, buf2, 2, 0);
    k_dwconv1d<<<eg, eb, 0, stream>>>(buf0, w14, b14, buf4, 2, 0);
    k_dwconv1d<<<eg, eb, 0, stream>>>(buf4, w12, b12, buf2, 0, 1);
    // ---- branch 2: out3 (D/H chains)
    k_dwconv333<<<eg, eb, 0, stream>>>(x2, wd, bd, buf0);
    k_dwconv1d<<<eg, eb, 0, stream>>>(buf0, w21, b21, buf4, 0, 0);
    k_dwconv1d<<<eg, eb, 0, stream>>>(buf4, w23, b23, buf3, 1, 0);
    k_dwconv1d<<<eg, eb, 0, stream>>>(buf0, w24, b24, buf4, 1, 0);
    k_dwconv1d<<<eg, eb, 0, stream>>>(buf4, w22, b22, buf3, 0, 1);

    // ---- instance norms (in place): buf1->out_k, buf2->out_q2, buf3->out_q1
    k_inorm_stats<<<128, 256, 0, stream>>>(buf1, stats);
    k_inorm_apply<<<eg, eb, 0, stream>>>(buf1, stats);
    k_inorm_stats<<<128, 256, 0, stream>>>(buf2, stats + 256);
    k_inorm_apply<<<eg, eb, 0, stream>>>(buf2, stats + 256);
    k_inorm_stats<<<128, 256, 0, stream>>>(buf3, stats + 512);
    k_inorm_apply<<<eg, eb, 0, stream>>>(buf3, stats + 512);

    // ---- pack QKV (order matters for buffer aliasing)
    dim3 qg(BATCH * HEADS * NTOK), qb(128);
    k_qkv_pack<<<qg, qb, 0, stream>>>(buf3, q1, 0, 1, 0);  // q1 <- out_q1 (buf3 read)
    k_qkv_pack<<<qg, qb, 0, stream>>>(buf1, k1, 0, 1, 0);  // k1 <- out_k
    k_qkv_pack<<<qg, qb, 0, stream>>>(buf1, v1t, 0, 0, 1); // v1ᵀ (overwrites buf3 lower half)
    k_qkv_pack<<<qg, qb, 0, stream>>>(buf2, q2, 1, 1, 0);  // q2 <- out_q2 (buf2 read)
    k_qkv_pack<<<qg, qb, 0, stream>>>(buf1, k2, 1, 1, 0);  // k2 (overwrites buf2 lower half)
    k_qkv_pack<<<qg, qb, 0, stream>>>(buf1, v2t, 1, 0, 1); // v2ᵀ

    // ---- attentions (fused softmax-divide + from_dh/from_dw + residual)
    dim3 ag(BATCH * HEADS * (NTOK / 16)), ab(256);
    k_attn<<<ag, ab, 0, stream>>>(q1, k1, v1t, buf1, buf4, 0);
    k_attn<<<ag, ab, 0, stream>>>(q2, k2, v2t, buf1, buf5, 1);

    // ---- final grouped conv over concat([o1, o2])
    k_finalconv<<<eg, eb, 0, stream>>>(buf4, buf5, wc, bc, out);
}
